// GCN_11398843203932
// MI455X (gfx1250) — matrix operand order repaired and checked by reading, not compiled, not run on hardware
//
#include <hip/hip_runtime.h>
#include <hip/hip_bf16.h>
#include <stdint.h>

#define NNODES 50000
#define MPAD   50016              // 1563 * 32, >= NNODES, multiple of 32
#define NEDGES 400000
#define BNEPS  1e-5f

typedef __attribute__((ext_vector_type(16))) __bf16 v16bf;
typedef __attribute__((ext_vector_type(8)))  float  v8f;

union FragAB { uint32_t u[8]; v16bf v; };

__device__ __forceinline__ uint32_t pack2_bf16(float a, float b) {
  uint32_t ua = __float_as_uint(a); ua = ua + 0x7fffu + ((ua >> 16) & 1u);
  uint32_t ub = __float_as_uint(b); ub = ub + 0x7fffu + ((ub >> 16) & 1u);
  return (ua >> 16) | (ub & 0xffff0000u);
}

__device__ __forceinline__ FragAB load_frag(const uint32_t* p) {
  uint4 lo = *(const uint4*)(p);
  uint4 hi = *(const uint4*)(p + 8);
  FragAB f;
  f.u[0] = lo.x; f.u[1] = lo.y; f.u[2] = lo.z; f.u[3] = lo.w;
  f.u[4] = hi.x; f.u[5] = hi.y; f.u[6] = hi.z; f.u[7] = hi.w;
  return f;
}

__device__ __forceinline__ FragAB load_frag_b(const uint32_t* p) {
  uint4 lo = *(const uint4*)(p);
  uint4 hi = *(const uint4*)(p + 4);
  FragAB f;
  f.u[0] = lo.x; f.u[1] = lo.y; f.u[2] = lo.z; f.u[3] = lo.w;
  f.u[4] = hi.x; f.u[5] = hi.y; f.u[6] = hi.z; f.u[7] = hi.w;
  return f;
}

#define WMMA_BF16(A, B, Cacc) \
  __builtin_amdgcn_wmma_f32_16x16x32_bf16(false, (A).v, false, (B).v, (short)0, (Cacc), false, false)

// ---------------- degree / norm ----------------
__global__ void edge_deg_kernel(const int* __restrict__ dst, float* __restrict__ deg) {
  int e = blockIdx.x * blockDim.x + threadIdx.x;
  if (e < NEDGES) atomicAdd(&deg[dst[e]], 1.0f);
}

__global__ void finish_dis_kernel(float* __restrict__ deg) {
  int n = blockIdx.x * blockDim.x + threadIdx.x;
  if (n < NNODES) deg[n] = rsqrtf(deg[n] + 1.0f);   // self loop included
}

// ---------------- f32 -> bf16 activation (K padded to Kp, M padded to MPAD) ----------------
__global__ void conv_bf16_kernel(const float* __restrict__ H, uint32_t* __restrict__ out,
                                 int Fi, int Kp) {
  size_t idx = (size_t)blockIdx.x * blockDim.x + threadIdx.x;
  size_t total = (size_t)MPAD * (Kp >> 1);
  if (idx >= total) return;
  int dwpr = Kp >> 1;
  int row = (int)(idx / dwpr);
  int k   = ((int)(idx % dwpr)) << 1;
  float a = 0.0f, b = 0.0f;
  if (row < NNODES) {
    if (k     < Fi) a = H[(size_t)row * Fi + k];
    if (k + 1 < Fi) b = H[(size_t)row * Fi + k + 1];
  }
  out[idx] = pack2_bf16(a, b);
}

// ---------------- weight repack into B-fragment tile layout ----------------
// out dword index = ((nt*KT + kt)*32 + lane)*8 + r
// lane: col n = nt*16 + (lane&15), half = lane>>4 ; VGPR r holds K = kt*32 + (r>>2)*16 + half*8 + 2(r&3), +1
__global__ void repack_w_kernel(const float* __restrict__ W, uint32_t* __restrict__ out,
                                int K, int Kp, int Nout) {
  int KT = Kp >> 5;
  int total = (Nout >> 4) * KT * 32 * 8;
  int idx = blockIdx.x * blockDim.x + threadIdx.x;
  if (idx >= total) return;
  int r    = idx & 7;
  int lane = (idx >> 3) & 31;
  int kt   = (idx >> 8) % KT;
  int nt   = (idx >> 8) / KT;
  int half = lane >> 4;
  int n    = nt * 16 + (lane & 15);
  int k0   = kt * 32 + ((r & 4) << 2) + half * 8 + 2 * (r & 3);
  float w0 = (k0     < K) ? W[(size_t)k0 * Nout + n]       : 0.0f;
  float w1 = (k0 + 1 < K) ? W[(size_t)(k0 + 1) * Nout + n] : 0.0f;
  out[idx] = pack2_bf16(w0, w1);
}

// ---------------- WMMA GEMM, 2x4 register-blocked: each wave -> 32x64 of C ----------------
__global__ __launch_bounds__(256)
void wmma_gemm2x4_kernel(const uint32_t* __restrict__ Abf,   // [MPAD, Kp/2] dwords
                         const uint32_t* __restrict__ Bpack,
                         float* __restrict__ C, int Kp, int Nout) {
  int wave = (int)((blockIdx.x * (size_t)blockDim.x + threadIdx.x) >> 5);
  int lane = threadIdx.x & 31;
  int mt2_cnt = MPAD >> 5, nt4_cnt = Nout >> 6;
  int tiles = mt2_cnt * nt4_cnt;
  if (wave >= tiles) return;                 // wave-uniform: EXEC stays all-ones
  int mt2 = wave % mt2_cnt, nt4 = wave / mt2_cnt;
  int half = lane >> 4, mlane = lane & 15;
  int KT = Kp >> 5;
  int dwpr = Kp >> 1;
  size_t bstride = (size_t)KT * 32 * 8;      // dwords per 16-wide N tile
  const uint32_t* arow0 = Abf + (size_t)(mt2 * 32 + mlane) * dwpr + half * 4;
  const uint32_t* arow1 = arow0 + (size_t)16 * dwpr;
  const uint32_t* bp0 = Bpack + (size_t)(4 * nt4) * bstride + (size_t)lane * 8;
  const uint32_t* bp1 = bp0 + bstride;
  const uint32_t* bp2 = bp0 + 2 * bstride;
  const uint32_t* bp3 = bp0 + 3 * bstride;
  v8f c00 = {}, c01 = {}, c02 = {}, c03 = {};
  v8f c10 = {}, c11 = {}, c12 = {}, c13 = {};
  for (int kt = 0; kt < KT; ++kt) {
    FragAB a0 = load_frag(arow0);
    FragAB a1 = load_frag(arow1);
    FragAB b0 = load_frag_b(bp0);
    FragAB b1 = load_frag_b(bp1);
    FragAB b2 = load_frag_b(bp2);
    FragAB b3 = load_frag_b(bp3);
    c00 = WMMA_BF16(a0, b0, c00);
    c10 = WMMA_BF16(a1, b0, c10);
    c01 = WMMA_BF16(a0, b1, c01);
    c11 = WMMA_BF16(a1, b1, c11);
    c02 = WMMA_BF16(a0, b2, c02);
    c12 = WMMA_BF16(a1, b2, c12);
    c03 = WMMA_BF16(a0, b3, c03);
    c13 = WMMA_BF16(a1, b3, c13);
    arow0 += 16; arow1 += 16;                // 32 K values = 16 dwords
    bp0 += 256; bp1 += 256; bp2 += 256; bp3 += 256;   // next K tile (32*8 dwords)
  }
  // store: tile(i,j): row = mt2*32 + i*16 + half*8 + r ; col = nt4*64 + j*16 + mlane
  float* base = C + (size_t)(mt2 * 32 + half * 8) * Nout + nt4 * 64 + mlane;
#pragma unroll
  for (int r = 0; r < 8; ++r) {
    float* r0 = base + (size_t)r * Nout;
    float* r1 = base + (size_t)(r + 16) * Nout;
    r0[0]  = c00[r]; r0[16] = c01[r]; r0[32] = c02[r]; r0[48] = c03[r];
    r1[0]  = c10[r]; r1[16] = c11[r]; r1[32] = c12[r]; r1[48] = c13[r];
  }
}

// ---------------- WMMA GEMM, 1x1 tile (for Nout == 16) ----------------
__global__ __launch_bounds__(256)
void wmma_gemm_kernel(const uint32_t* __restrict__ Abf, const uint32_t* __restrict__ Bpack,
                      float* __restrict__ C, int Kp, int Nout) {
  int wave = (int)((blockIdx.x * (size_t)blockDim.x + threadIdx.x) >> 5);
  int lane = threadIdx.x & 31;
  int mt_cnt = MPAD >> 4, nt_cnt = Nout >> 4;
  int tiles = mt_cnt * nt_cnt;
  if (wave >= tiles) return;
  int mt = wave % mt_cnt, nt = wave / mt_cnt;
  int half = lane >> 4, mlane = lane & 15;
  int KT = Kp >> 5;
  const uint32_t* arow = Abf + (size_t)(mt * 16 + mlane) * (Kp >> 1) + half * 4;
  const uint32_t* bptr = Bpack + ((size_t)(nt * KT) * 32 + lane) * 8;
  v8f c = {};
  for (int kt = 0; kt < KT; ++kt) {
    FragAB fa = load_frag(arow);
    FragAB fb = load_frag_b(bptr);
    c = WMMA_BF16(fa, fb, c);
    arow += 16;
    bptr += 256;
  }
  float* crow = C + (size_t)(mt * 16 + half * 8) * Nout + nt * 16 + mlane;
#pragma unroll
  for (int r = 0; r < 8; ++r) crow[(size_t)r * Nout] = c[r];
}

// ---------------- self-loop init: agg = hw * dis^2 ----------------
__global__ void selfloop_kernel(const float* __restrict__ hw, const float* __restrict__ dis,
                                float* __restrict__ agg, int Fo) {
  size_t i = (size_t)blockIdx.x * blockDim.x + threadIdx.x;
  size_t total = (size_t)NNODES * Fo;
  if (i >= total) return;
  int n = (int)(i / Fo);
  float d = dis[n];
  agg[i] = hw[i] * d * d;
}

// ---------------- edge scatter: agg[dst] += hw[src] * dis[src]*dis[dst] ----------------
__global__ void edge_scatter_kernel(const int* __restrict__ src, const int* __restrict__ dst,
                                    const float* __restrict__ dis, const float* __restrict__ hw,
                                    float* __restrict__ agg, int Fo) {
  int e = blockIdx.x;
  int s = src[e], d = dst[e];
  float nrm = dis[s] * dis[d];
  const float* hs = hw + (size_t)s * Fo;
  float* ad = agg + (size_t)d * Fo;
  for (int f = threadIdx.x; f < Fo; f += blockDim.x)
    atomicAdd(&ad[f], hs[f] * nrm);
}

// ---------------- bias + relu + column-stat partials ----------------
__global__ void bias_relu_stats_kernel(const float* __restrict__ agg, const float* __restrict__ bias,
                                       float* __restrict__ h_out, float* __restrict__ colsum,
                                       float* __restrict__ colsq, int Fo, int rows_per_chunk) {
  int f  = blockIdx.x * blockDim.x + threadIdx.x;   // Fo multiple of 256 here
  int r0 = blockIdx.y * rows_per_chunk;
  int r1 = r0 + rows_per_chunk; if (r1 > NNODES) r1 = NNODES;
  float b = bias[f], s = 0.0f, s2 = 0.0f;
  for (int r = r0; r < r1; ++r) {
    float v = agg[(size_t)r * Fo + f] + b;
    v = v > 0.0f ? v : 0.0f;
    h_out[(size_t)r * Fo + f] = v;
    s += v; s2 += v * v;
  }
  atomicAdd(&colsum[f], s);
  atomicAdd(&colsq[f], s2);
}

// ---------------- BatchNorm apply (training mode, biased var) ----------------
__global__ void bn_apply_kernel(float* __restrict__ h, const float* __restrict__ colsum,
                                const float* __restrict__ colsq, const float* __restrict__ g,
                                const float* __restrict__ be, int Fo) {
  size_t i = (size_t)blockIdx.x * blockDim.x + threadIdx.x;
  size_t total = (size_t)NNODES * Fo;
  if (i >= total) return;
  int f = (int)(i % Fo);
  float inv = 1.0f / (float)NNODES;
  float mean = colsum[f] * inv;
  float var  = colsq[f] * inv - mean * mean;
  h[i] = (h[i] - mean) * rsqrtf(var + BNEPS) * g[f] + be[f];
}

// ---------------- simple bias + relu (FC head) ----------------
__global__ void bias_relu_kernel(float* __restrict__ h, const float* __restrict__ b,
                                 size_t total, int F) {
  size_t i = (size_t)blockIdx.x * blockDim.x + threadIdx.x;
  if (i >= total) return;
  float v = h[i] + b[i % F];
  h[i] = v > 0.0f ? v : 0.0f;
}

// ---------------- final 16 -> 1 ----------------
__global__ void fc4_kernel(const float* __restrict__ h, const float* __restrict__ W,
                           const float* __restrict__ b, float* __restrict__ out) {
  int n = blockIdx.x * blockDim.x + threadIdx.x;
  if (n >= NNODES) return;
  float s = b[0];
#pragma unroll
  for (int k = 0; k < 16; ++k) s += h[(size_t)n * 16 + k] * W[k];
  out[n] = s;
}

extern "C" void kernel_launch(void* const* d_in, const int* in_sizes, int n_in,
                              void* d_out, int out_size, void* d_ws, size_t ws_size,
                              hipStream_t stream) {
  (void)in_sizes; (void)n_in; (void)out_size; (void)ws_size;
  const float* x  = (const float*)d_in[0];
  const int*   ei = (const int*)d_in[1];
  const int* src = ei;
  const int* dst = ei + NEDGES;
  const float *W[5], *bia[5], *gam[5], *bet[5];
  for (int i = 0; i < 5; ++i) {
    W[i]   = (const float*)d_in[2 + 4 * i];
    bia[i] = (const float*)d_in[3 + 4 * i];
    gam[i] = (const float*)d_in[4 + 4 * i];
    bet[i] = (const float*)d_in[5 + 4 * i];
  }
  const float* fcW2 = (const float*)d_in[22]; const float* fcb2 = (const float*)d_in[23];
  const float* fcW3 = (const float*)d_in[24]; const float* fcb3 = (const float*)d_in[25];
  const float* fcW4 = (const float*)d_in[26]; const float* fcb4 = (const float*)d_in[27];

  char* ws = (char*)d_ws;
  size_t off = 0;
  auto carve = [&](size_t bytes) -> char* {
    char* p = ws + off;
    off = (off + bytes + 255) & ~(size_t)255;
    return p;
  };
  float*    dis   = (float*)carve((size_t)NNODES * 4);
  uint32_t* hbf   = (uint32_t*)carve((size_t)MPAD * 1024 * 2);
  uint32_t* wpack = (uint32_t*)carve((size_t)2 * 1024 * 1024);
  float* bufs[3];
  bufs[0] = (float*)carve((size_t)MPAD * 1024 * 4);
  bufs[1] = (float*)carve((size_t)MPAD * 1024 * 4);
  bufs[2] = (float*)carve((size_t)MPAD * 1024 * 4);
  float* colsum = (float*)carve(2048 * 4);
  float* colsq  = colsum + 1024;

  // degrees -> dis = rsqrt(deg+1)
  hipMemsetAsync(dis, 0, (size_t)NNODES * 4, stream);
  edge_deg_kernel<<<(NEDGES + 255) / 256, 256, 0, stream>>>(dst, dis);
  finish_dis_kernel<<<(NNODES + 255) / 256, 256, 0, stream>>>(dis);

  const int Fis[5] = {30, 1024, 512, 256, 512};
  const int Fos[5] = {1024, 512, 256, 512, 1024};
  const float* hin = x;
  for (int i = 0; i < 5; ++i) {
    int Fi = Fis[i], Fo = Fos[i];
    int Kp = (Fi + 31) & ~31;
    float* hw  = bufs[i % 3];
    float* agg = bufs[(i + 1) % 3];

    size_t cvt_total = (size_t)MPAD * (Kp >> 1);
    conv_bf16_kernel<<<(unsigned)((cvt_total + 255) / 256), 256, 0, stream>>>(hin, hbf, Fi, Kp);
    int rp_total = (Fo >> 4) * (Kp >> 5) * 256;
    repack_w_kernel<<<(rp_total + 255) / 256, 256, 0, stream>>>(W[i], wpack, Fi, Kp, Fo);

    size_t tiles = (size_t)(MPAD / 32) * (Fo / 64);
    wmma_gemm2x4_kernel<<<(unsigned)((tiles + 7) / 8), 256, 0, stream>>>(hbf, wpack, hw, Kp, Fo);

    size_t ew_total = (size_t)NNODES * Fo;
    selfloop_kernel<<<(unsigned)((ew_total + 255) / 256), 256, 0, stream>>>(hw, dis, agg, Fo);
    edge_scatter_kernel<<<NEDGES, 256, 0, stream>>>(src, dst, dis, hw, agg, Fo);

    hipMemsetAsync(colsum, 0, 2048 * 4, stream);
    dim3 sgrid(Fo / 256, (NNODES + 255) / 256);
    bias_relu_stats_kernel<<<sgrid, 256, 0, stream>>>(agg, bia[i], hw, colsum, colsq, Fo, 256);
    bn_apply_kernel<<<(unsigned)((ew_total + 255) / 256), 256, 0, stream>>>(hw, colsum, colsq, gam[i], bet[i], Fo);
    hin = hw;
  }

  // FC2: 1024 -> 128, relu        (hin == bufs[1])
  {
    float* out = bufs[2];
    size_t cvt_total = (size_t)MPAD * (1024 >> 1);
    conv_bf16_kernel<<<(unsigned)((cvt_total + 255) / 256), 256, 0, stream>>>(hin, hbf, 1024, 1024);
    int rp_total = (128 >> 4) * (1024 >> 5) * 256;
    repack_w_kernel<<<(rp_total + 255) / 256, 256, 0, stream>>>(fcW2, wpack, 1024, 1024, 128);
    size_t tiles = (size_t)(MPAD / 32) * (128 / 64);
    wmma_gemm2x4_kernel<<<(unsigned)((tiles + 7) / 8), 256, 0, stream>>>(hbf, wpack, out, 1024, 128);
    size_t tot = (size_t)NNODES * 128;
    bias_relu_kernel<<<(unsigned)((tot + 255) / 256), 256, 0, stream>>>(out, fcb2, tot, 128);
    hin = out;
  }
  // FC3: 128 -> 16, relu          (hin == bufs[2])
  {
    float* out = bufs[0];
    size_t cvt_total = (size_t)MPAD * (128 >> 1);
    conv_bf16_kernel<<<(unsigned)((cvt_total + 255) / 256), 256, 0, stream>>>(hin, hbf, 128, 128);
    int rp_total = (16 >> 4) * (128 >> 5) * 256;
    repack_w_kernel<<<(rp_total + 255) / 256, 256, 0, stream>>>(fcW3, wpack, 128, 128, 16);
    size_t tiles = (size_t)(MPAD / 16) * (16 / 16);
    wmma_gemm_kernel<<<(unsigned)((tiles + 7) / 8), 256, 0, stream>>>(hbf, wpack, out, 128, 16);
    size_t tot = (size_t)NNODES * 16;
    bias_relu_kernel<<<(unsigned)((tot + 255) / 256), 256, 0, stream>>>(out, fcb3, tot, 16);
    hin = out;
  }
  // FC4: 16 -> 1
  fc4_kernel<<<(NNODES + 255) / 256, 256, 0, stream>>>(hin, fcW4, fcb4, (float*)d_out);
}
